// Net_GCN_36335423324385
// MI455X (gfx1250) — compile-verified, matrix operand order checked by reading, not run on hardware
//
#include <hip/hip_runtime.h>
#include <hip/hip_bf16.h>

typedef __attribute__((ext_vector_type(16))) __bf16 v16bf;
typedef __attribute__((ext_vector_type(8)))  float  v8f;

__device__ __forceinline__ __bf16 bf(float f) { return (__bf16)f; }

// ---------------------------------------------------------------------------
// C[M,N](ldc) = act(A[M,K](lda) @ B[K,N] + bias)
// Requirements (guaranteed by caller): K % 32 == 0, M % 16 == 0, N % 16 == 0.
// One wave32 computes one 16x16 tile via v_wmma_f32_16x16x32_bf16 (f32 accum).
// ACT: 0 = none, 1 = relu, 2 = sigmoid (compile-time; branch-free epilogue).
// ---------------------------------------------------------------------------
template <int ACT, bool HAS_BIAS>
__global__ __launch_bounds__(256) void gemm_bf16_wmma(
    const float* __restrict__ A, int lda,
    const float* __restrict__ B,
    const float* __restrict__ bias,
    float* __restrict__ C, int ldc,
    int M16, int N, int K) {
  const int wave = threadIdx.x >> 5;
  const int lane = threadIdx.x & 31;
  const int h    = lane >> 4;            // half-wave: K-subgroup select
  const int l    = lane & 15;
  const int tm   = blockIdx.x * 8 + wave;
  if (tm >= M16) return;                 // wave-uniform early out
  const int  cn   = blockIdx.y * 16;
  const long arow = (long)(tm * 16 + l) * lda;

  v8f acc = {};
  for (int kk = 0; kk < K; kk += 32) {
    const int k0 = kk + 8 * h;
    // A fragment: elements 0-7 <-> k0..k0+7, elements 8-15 <-> k0+16..k0+23
    const float4 a0 = *(const float4*)(A + arow + k0);
    const float4 a1 = *(const float4*)(A + arow + k0 + 4);
    const float4 a2 = *(const float4*)(A + arow + k0 + 16);
    const float4 a3 = *(const float4*)(A + arow + k0 + 20);
    v16bf ab, bb;
    ab[0]  = bf(a0.x); ab[1]  = bf(a0.y); ab[2]  = bf(a0.z); ab[3]  = bf(a0.w);
    ab[4]  = bf(a1.x); ab[5]  = bf(a1.y); ab[6]  = bf(a1.z); ab[7]  = bf(a1.w);
    ab[8]  = bf(a2.x); ab[9]  = bf(a2.y); ab[10] = bf(a2.z); ab[11] = bf(a2.w);
    ab[12] = bf(a3.x); ab[13] = bf(a3.y); ab[14] = bf(a3.z); ab[15] = bf(a3.w);
    // B fragment: lane l holds column cn+l; rows follow same K pattern
    const float* bp = B + (long)k0 * N + (cn + l);
#pragma unroll
    for (int e = 0; e < 8; ++e) bb[e] = bf(bp[(long)e * N]);
#pragma unroll
    for (int e = 0; e < 8; ++e) bb[8 + e] = bf(bp[(long)(16 + e) * N]);

    acc = __builtin_amdgcn_wmma_f32_16x16x32_bf16(
        false, ab, false, bb, (short)0, acc, false, false);
  }

  float bv = 0.0f;
  if constexpr (HAS_BIAS) bv = bias[cn + l];
#pragma unroll
  for (int j = 0; j < 8; ++j) {
    const int row = tm * 16 + j + 8 * h;   // C/D layout: VGPR j, half h
    float v = acc[j] + bv;
    if constexpr (ACT == 1) v = fmaxf(v, 0.0f);
    if constexpr (ACT == 2) v = __fdividef(1.0f, 1.0f + __expf(-v));
    C[(long)row * ldc + (cn + l)] = v;
  }
}

// ---------------------------------------------------------------------------
// Index narrowing, degree / normalization
// ---------------------------------------------------------------------------
__global__ void k_cvt_idx(const long long* __restrict__ in, int* __restrict__ o,
                          long long n) {
  long long i = (long long)blockIdx.x * blockDim.x + threadIdx.x;
  if (i < n) o[i] = (int)in[i];
}
__global__ void k_set_ones(float* p, int n) {
  int i = blockIdx.x * blockDim.x + threadIdx.x;
  if (i < n) p[i] = 1.0f;                 // self-loop contributes 1 to degree
}
__global__ void k_deg_accum(const int* __restrict__ dst, float* deg,
                            long long E) {
  long long e = (long long)blockIdx.x * blockDim.x + threadIdx.x;
  if (e < E) atomicAdd(&deg[dst[e]], 1.0f);
}
__global__ void k_rsqrt(float* p, int n) {
  int i = blockIdx.x * blockDim.x + threadIdx.x;
  if (i < n) p[i] = rsqrtf(p[i]);         // deg >= 1 always (self-loop)
}
__global__ void k_pad_w2(const float* __restrict__ W2, float* __restrict__ Wp) {
  int i = blockIdx.x * blockDim.x + threadIdx.x;   // 32x32 padded, rows>=16 zero
  if (i < 32 * 32) Wp[i] = (i < 16 * 32) ? W2[i] : 0.0f;
}

// ---------------------------------------------------------------------------
// Aggregation:  O[d,:] = sum_{e:dst=d} H[src]*dinv[src]*dinv[dst]
//               + H[d]*dinv[d]^2 (self loop); then +bias, relu.
// O has row stride ldo >= F; pad columns are zero-filled by k_agg_self.
// ---------------------------------------------------------------------------
__global__ void k_agg_self(const float* __restrict__ H,
                           const float* __restrict__ dinv,
                           float* __restrict__ O, int N, int F, int ldo) {
  long long idx = (long long)blockIdx.x * blockDim.x + threadIdx.x;
  long long total = (long long)N * ldo;
  if (idx < total) {
    int i = (int)(idx / ldo), f = (int)(idx % ldo);
    float di = dinv[i];
    O[idx] = (f < F) ? H[(long)i * F + f] * di * di : 0.0f;
  }
}
__global__ void k_agg_edges(const float* __restrict__ H,
                            const float* __restrict__ dinv,
                            const int* __restrict__ src,
                            const int* __restrict__ dst,
                            float* __restrict__ O, long long E, int F, int ldo) {
  long long e = (long long)blockIdx.x * blockDim.x + threadIdx.x;
  if (e >= E) return;
  const int c = blockIdx.y * 4;           // 4-feature chunk
  const int s = src[e], d = dst[e];
  const float nrm = dinv[s] * dinv[d];
  const float4 hv = *(const float4*)(H + (long)s * F + c);
  float* o = O + (long)d * ldo + c;
  atomicAdd(o + 0, hv.x * nrm);
  atomicAdd(o + 1, hv.y * nrm);
  atomicAdd(o + 2, hv.z * nrm);
  atomicAdd(o + 3, hv.w * nrm);
}
__global__ void k_bias_relu(float* __restrict__ O, const float* __restrict__ b,
                            int N, int F, int ldo) {
  long long idx = (long long)blockIdx.x * blockDim.x + threadIdx.x;
  long long total = (long long)N * F;
  if (idx < total) {
    int i = (int)(idx / F), f = (int)(idx % F);
    float* p = O + (long)i * ldo + f;
    float v = *p + b[f];
    *p = v > 0.0f ? v : 0.0f;
  }
}

// ---------------------------------------------------------------------------
// Global max pool (F = 64). Post-relu values >= 0 so uint bit-order == float
// order; init-to-0 matches reference's empty-segment -> 0 semantics.
// ---------------------------------------------------------------------------
__global__ void k_zero(float* p, int n) {
  int i = blockIdx.x * blockDim.x + threadIdx.x;
  if (i < n) p[i] = 0.0f;
}
__global__ void k_pool_max(const float* __restrict__ H,
                           const long long* __restrict__ batch,
                           float* __restrict__ P, long long total) {
  long long idx = (long long)blockIdx.x * blockDim.x + threadIdx.x;
  if (idx < total) {
    int i = (int)(idx >> 6), f = (int)(idx & 63);
    unsigned v = __float_as_uint(H[idx]);
    atomicMax((unsigned*)&P[((long long)batch[i] << 6) + f], v);
  }
}

// ---------------------------------------------------------------------------
extern "C" void kernel_launch(void* const* d_in, const int* in_sizes, int n_in,
                              void* d_out, int out_size, void* d_ws, size_t ws_size,
                              hipStream_t stream) {
  const float*     x     = (const float*)d_in[0];
  const long long* ei    = (const long long*)d_in[1];   // int64 per reference
  const long long* batch = (const long long*)d_in[2];
  const float* W1 = (const float*)d_in[3];  const float* b1 = (const float*)d_in[4];
  const float* W2 = (const float*)d_in[5];  const float* b2 = (const float*)d_in[6];
  const float* W3 = (const float*)d_in[7];  const float* b3 = (const float*)d_in[8];
  const float* L1w = (const float*)d_in[9];  const float* L1b = (const float*)d_in[10];
  const float* L2w = (const float*)d_in[11]; const float* L2b = (const float*)d_in[12];
  float* out = (float*)d_out;

  const int       N   = in_sizes[2];          // 100000  (multiple of 16)
  const long long E   = in_sizes[1] / 2;      // 1600000
  const int       FIN = in_sizes[0] / N;      // 128
  const int       G   = 1024;
  const int       M16 = N / 16;

  // workspace carve-out (~66 MB)
  float* dinv   = (float*)d_ws;                 // N
  int*   idx32  = (int*)(dinv + N);             // 2E
  float* Wpad   = (float*)(idx32 + 2 * E);      // 32*32 zero-padded W2
  float* H      = Wpad + 1024;                  // N x 64 max
  float* AGG    = H + (size_t)N * 64;           // N x 64 max (padded strides)
  float* pooled = AGG + (size_t)N * 64;         // G x 64
  float* z      = pooled + (size_t)G * 64;      // G x 256

  const int* src = idx32;
  const int* dst = idx32 + E;

  const int T = 256;
  auto blks = [](long long n, int t) { return (unsigned)((n + t - 1) / t); };

  // --- index narrowing + symmetric normalization coefficients ---
  k_cvt_idx<<<blks(2 * E, T), T, 0, stream>>>(ei, idx32, 2 * E);
  k_set_ones<<<blks(N, T), T, 0, stream>>>(dinv, N);
  k_deg_accum<<<blks(E, T), T, 0, stream>>>(dst, dinv, E);
  k_rsqrt<<<blks(N, T), T, 0, stream>>>(dinv, N);
  k_pad_w2<<<4, T, 0, stream>>>(W2, Wpad);

  // --- one GCN conv layer: gemm -> self term (+zero pad) -> edge scatter ---
  auto conv = [&](const float* in, int lda, int K, const float* Wt,
                  const float* bs, int Fout, int ldo) {
    dim3 gg((M16 + 7) / 8, Fout / 16);
    gemm_bf16_wmma<0, false><<<gg, T, 0, stream>>>(in, lda, Wt, nullptr, H,
                                                   Fout, M16, Fout, K);
    k_agg_self<<<blks((long long)N * ldo, T), T, 0, stream>>>(H, dinv, AGG,
                                                              N, Fout, ldo);
    dim3 ge(blks(E, T), Fout / 4);
    k_agg_edges<<<ge, T, 0, stream>>>(H, dinv, src, dst, AGG, E, Fout, ldo);
    k_bias_relu<<<blks((long long)N * Fout, T), T, 0, stream>>>(AGG, bs,
                                                                N, Fout, ldo);
  };

  conv(x,   FIN, FIN, W1,   b1, 16, 32);   // pad cols 16..31 of AGG to zero
  conv(AGG, 32,  32,  Wpad, b2, 32, 32);   // zero-padded K and W2
  conv(AGG, 32,  32,  W3,   b3, 64, 64);

  // --- global max pool ---
  k_zero<<<blks(G * 64, T), T, 0, stream>>>(pooled, G * 64);
  k_pool_max<<<blks((long long)N * 64, T), T, 0, stream>>>(
      AGG, batch, pooled, (long long)N * 64);

  // --- MLP head ---
  dim3 g1((G / 16 + 7) / 8, 256 / 16);
  gemm_bf16_wmma<1, true><<<g1, T, 0, stream>>>(pooled, 64, L1w, L1b, z, 256,
                                                G / 16, 256, 64);
  gemm_bf16_wmma<2, true><<<g1, T, 0, stream>>>(z, 256, L2w, L2b, out, 256,
                                                G / 16, 256, 256);
}
